// GCNModel_15212774162444
// MI455X (gfx1250) — matrix operand order repaired and checked by reading, not compiled, not run on hardware
//
#include <hip/hip_runtime.h>
#include <hip/hip_bf16.h>

#define GCN_DIN  128
#define GCN_DH   128
#define GCN_DOUT 32
#define GCN_EPS  1e-5f

typedef __attribute__((ext_vector_type(16))) __bf16 v16bf;
typedef __attribute__((ext_vector_type(8)))  float  v8f;

// ---------------------------------------------------------------------------
// WMMA GEMM: T[n_rows, NOUT] = X[n_rows, 128] @ W[128, NOUT]   (bf16 in, f32 acc)
// One wave computes a 16 x NOUT strip; 8 waves (256 threads) = 128 rows/block.
// W staged in LDS pre-swizzled into per-lane B-fragment layout:
//   B fragment (32x16 bf16): lane L -> n = L%16, element e -> k = e + 16*(L/16)
// A fragment (16x32 bf16): lane L -> m = L%16, half = L/16,
//   element pair (v,p): k = 2v+p + 16*half
// C/D (16x16 f32): lane L -> n = L%16, vgpr r -> m = r + 8*(L/16)
// ---------------------------------------------------------------------------
template <int NOUT>
__global__ __launch_bounds__(256) void gcn_gemm_wmma(
    const float* __restrict__ X, const float* __restrict__ W,
    float* __restrict__ T, int n_rows) {
  constexpr int KTILES = 128 / 32;   // 4
  constexpr int NTILES = NOUT / 16;  // 8 or 2
  __shared__ __align__(32) __bf16 Bfrag[KTILES][NTILES][32][16];

  const int tid = threadIdx.x;
  constexpr int TOTAL = KTILES * NTILES * 32 * 16;
  for (int f = tid; f < TOTAL; f += 256) {
    const int kt   = f / (NTILES * 512);
    const int rem  = f % (NTILES * 512);
    const int nt   = rem / 512;
    const int lane = (rem % 512) / 16;
    const int e    = f % 16;
    const int k    = kt * 32 + e + 16 * (lane / 16);
    const int n    = nt * 16 + (lane % 16);
    Bfrag[kt][nt][lane][e] = (__bf16)W[k * NOUT + n];
  }
  __syncthreads();

  const int wave = tid / 32;
  const int lane = tid % 32;
  const int half = lane / 16;
  const int m16  = lane % 16;

  const long long rowBase = (long long)blockIdx.x * 128 + (long long)wave * 16;
  long long rowA = rowBase + m16;
  if (rowA >= n_rows) rowA = n_rows - 1;  // clamp loads; stores are guarded

  v8f acc[NTILES] = {};

#pragma unroll
  for (int kt = 0; kt < KTILES; ++kt) {
    v16bf a;
    const float* xrow = X + rowA * 128 + kt * 32 + half * 16;
#pragma unroll
    for (int v = 0; v < 8; ++v) {
      const int kb = 2 * v;
      a[2 * v]     = (__bf16)xrow[kb];
      a[2 * v + 1] = (__bf16)xrow[kb + 1];
    }
#pragma unroll
    for (int nt = 0; nt < NTILES; ++nt) {
      v16bf b = *(const v16bf*)&Bfrag[kt][nt][lane][0];
      acc[nt] = __builtin_amdgcn_wmma_f32_16x16x32_bf16(
          false, a, false, b, (short)0, acc[nt], false, false);
    }
  }

#pragma unroll
  for (int nt = 0; nt < NTILES; ++nt) {
#pragma unroll
    for (int r = 0; r < 8; ++r) {
      const long long row = rowBase + r + 8 * half;
      if (row < n_rows) T[row * NOUT + nt * 16 + m16] = acc[nt][r];
    }
  }
}

// ---------------------------------------------------------------------------
// Degree / normalization
// ---------------------------------------------------------------------------
__global__ void deg_init(float* deg, int n) {
  int i = blockIdx.x * blockDim.x + threadIdx.x;
  if (i < n) deg[i] = 1.0f;  // self loop
}

__global__ void deg_accum(const long long* __restrict__ ei, float* deg, int E) {
  int e = blockIdx.x * blockDim.x + threadIdx.x;
  if (e < E) atomicAdd(&deg[ei[E + e]], 1.0f);  // dst row
}

__global__ void deg_to_dinv(float* deg, int n) {
  int i = blockIdx.x * blockDim.x + threadIdx.x;
  if (i < n) deg[i] = rsqrtf(deg[i]);  // deg >= 1 always (self loops)
}

// out[i, 4j..4j+3] = bias + dinv[i]^2 * T[i, ...]   (self-loop + bias), float4
template <int D>
__global__ void init_out4(const float* __restrict__ T, const float* __restrict__ bias,
                          const float* __restrict__ dinv, float* __restrict__ out, int n) {
  constexpr int DV = D / 4;
  long long idx = (long long)blockIdx.x * blockDim.x + threadIdx.x;
  long long total = (long long)n * DV;
  if (idx >= total) return;
  const int j = (int)(idx % DV);
  const long long node = idx / DV;
  const float dv = dinv[node];
  const float w = dv * dv;
  const float4 t = ((const float4*)T)[idx];
  const float4 b = ((const float4*)bias)[j];
  float4 o;
  o.x = b.x + w * t.x; o.y = b.y + w * t.y;
  o.z = b.z + w * t.z; o.w = b.w + w * t.w;
  ((float4*)out)[idx] = o;
}

// Wave-per-edge scatter, D=128: lane moves one float4 (512B coalesced per wave)
__global__ __launch_bounds__(256) void edge_scatter128(
    const float* __restrict__ T, const long long* __restrict__ ei,
    const float* __restrict__ dinv, float* __restrict__ out, int E) {
  long long gid = (long long)blockIdx.x * blockDim.x + threadIdx.x;
  const int e = (int)(gid >> 5);
  const int lane = (int)(gid & 31);
  if (e >= E) return;  // uniform across the wave
  const long long s = ei[e];
  const long long d = ei[E + e];
  const float w = dinv[s] * dinv[d];
  const float4 v = *(const float4*)(T + s * 128 + lane * 4);
  float* o = out + d * 128 + lane * 4;
  atomicAdd(o + 0, w * v.x);
  atomicAdd(o + 1, w * v.y);
  atomicAdd(o + 2, w * v.z);
  atomicAdd(o + 3, w * v.w);
}

// Wave-per-edge scatter, D=32: lane moves one float
__global__ __launch_bounds__(256) void edge_scatter32(
    const float* __restrict__ T, const long long* __restrict__ ei,
    const float* __restrict__ dinv, float* __restrict__ out, int E) {
  long long gid = (long long)blockIdx.x * blockDim.x + threadIdx.x;
  const int e = (int)(gid >> 5);
  const int lane = (int)(gid & 31);
  if (e >= E) return;
  const long long s = ei[e];
  const long long d = ei[E + e];
  const float w = dinv[s] * dinv[d];
  atomicAdd(out + d * 32 + lane, w * T[s * 32 + lane]);
}

// ---------------------------------------------------------------------------
// BatchNorm (training mode, biased variance) + ReLU — coalesced float4 pass
// ---------------------------------------------------------------------------
__global__ void stats_zero(float* p, int cnt) {
  int i = blockIdx.x * blockDim.x + threadIdx.x;
  if (i < cnt) p[i] = 0.f;
}

// B is [n,128] f32 viewed as n*32 float4's. Thread's channel group j = idx%32
// is stride-invariant (stride % 32 == 0). LDS-reduce per block, then atomics.
__global__ __launch_bounds__(256) void bn_partial128(
    const float* __restrict__ B, float* __restrict__ sum, float* __restrict__ sumsq,
    long long nvec) {
  __shared__ float ls[128], ls2[128];
  if (threadIdx.x < 128) { ls[threadIdx.x] = 0.f; ls2[threadIdx.x] = 0.f; }
  __syncthreads();

  const long long stride = (long long)gridDim.x * blockDim.x;  // multiple of 32
  const long long idx0 = (long long)blockIdx.x * blockDim.x + threadIdx.x;
  float4 s = {0.f, 0.f, 0.f, 0.f}, s2 = {0.f, 0.f, 0.f, 0.f};
  for (long long f = idx0; f < nvec; f += stride) {
    const float4 v = ((const float4*)B)[f];
    s.x += v.x; s.y += v.y; s.z += v.z; s.w += v.w;
    s2.x += v.x * v.x; s2.y += v.y * v.y; s2.z += v.z * v.z; s2.w += v.w * v.w;
  }
  const int j = (int)(idx0 & 31) * 4;
  atomicAdd(&ls[j + 0], s.x);  atomicAdd(&ls2[j + 0], s2.x);
  atomicAdd(&ls[j + 1], s.y);  atomicAdd(&ls2[j + 1], s2.y);
  atomicAdd(&ls[j + 2], s.z);  atomicAdd(&ls2[j + 2], s2.z);
  atomicAdd(&ls[j + 3], s.w);  atomicAdd(&ls2[j + 3], s2.w);
  __syncthreads();
  if (threadIdx.x < 128) {
    atomicAdd(&sum[threadIdx.x], ls[threadIdx.x]);
    atomicAdd(&sumsq[threadIdx.x], ls2[threadIdx.x]);
  }
}

__global__ void bn_finalize(float* mean, float* var, int n, int D) {
  int c = blockIdx.x * blockDim.x + threadIdx.x;
  if (c < D) {
    const float m = mean[c] / (float)n;
    mean[c] = m;
    var[c] = var[c] / (float)n - m * m;
  }
}

template <int D>
__global__ void bn_relu4(const float* __restrict__ B, const float* __restrict__ mean,
                         const float* __restrict__ var, const float* __restrict__ g,
                         const float* __restrict__ be, float* __restrict__ C, int n) {
  constexpr int DV = D / 4;
  long long idx = (long long)blockIdx.x * blockDim.x + threadIdx.x;
  long long total = (long long)n * DV;
  if (idx >= total) return;
  const int j = (int)(idx % DV);
  const float4 v = ((const float4*)B)[idx];
  const float4 m = ((const float4*)mean)[j];
  const float4 va = ((const float4*)var)[j];
  const float4 gg = ((const float4*)g)[j];
  const float4 bb = ((const float4*)be)[j];
  float4 o;
  o.x = (v.x - m.x) * rsqrtf(va.x + GCN_EPS) * gg.x + bb.x;
  o.y = (v.y - m.y) * rsqrtf(va.y + GCN_EPS) * gg.y + bb.y;
  o.z = (v.z - m.z) * rsqrtf(va.z + GCN_EPS) * gg.z + bb.z;
  o.w = (v.w - m.w) * rsqrtf(va.w + GCN_EPS) * gg.w + bb.w;
  o.x = o.x > 0.f ? o.x : 0.f;
  o.y = o.y > 0.f ? o.y : 0.f;
  o.z = o.z > 0.f ? o.z : 0.f;
  o.w = o.w > 0.f ? o.w : 0.f;
  ((float4*)C)[idx] = o;
}

// ---------------------------------------------------------------------------
// Row-wise log_softmax over 32 columns: one wave32 per row.
// ---------------------------------------------------------------------------
__global__ __launch_bounds__(256) void log_softmax32(const float* __restrict__ B,
                                                     float* __restrict__ out, int n) {
  const int lane = threadIdx.x % 32;
  const int row = blockIdx.x * (blockDim.x / 32) + threadIdx.x / 32;
  if (row >= n) return;  // uniform across the wave
  const float v = B[(long long)row * 32 + lane];
  float mx = v;
#pragma unroll
  for (int o = 16; o > 0; o >>= 1) mx = fmaxf(mx, __shfl_xor(mx, o, 32));
  float s = expf(v - mx);
#pragma unroll
  for (int o = 16; o > 0; o >>= 1) s += __shfl_xor(s, o, 32);
  out[(long long)row * 32 + lane] = v - mx - logf(s);
}

// ---------------------------------------------------------------------------
extern "C" void kernel_launch(void* const* d_in, const int* in_sizes, int n_in,
                              void* d_out, int out_size, void* d_ws, size_t ws_size,
                              hipStream_t stream) {
  const float*     x   = (const float*)d_in[0];
  const long long* ei  = (const long long*)d_in[1];
  const float*     W0  = (const float*)d_in[2];
  const float*     b0  = (const float*)d_in[3];
  const float*     g0  = (const float*)d_in[4];
  const float*     be0 = (const float*)d_in[5];
  const float*     W1  = (const float*)d_in[6];
  const float*     b1  = (const float*)d_in[7];
  const float*     g1  = (const float*)d_in[8];
  const float*     be1 = (const float*)d_in[9];
  const float*     W2  = (const float*)d_in[10];
  const float*     b2  = (const float*)d_in[11];

  const int n = in_sizes[0] / GCN_DIN;  // 100000
  const int E = in_sizes[1] / 2;        // 1600000
  float* out = (float*)d_out;

  // Workspace carve-out (floats); all sub-buffers stay 16B-aligned.
  float* ws   = (float*)d_ws;
  float* dinv = ws;                                   // n
  float* mean = dinv + ((n + 3) & ~3);                // 128
  float* var  = mean + 128;                           // 128
  float* bufT = var + 128;                            // n*128
  float* bufB = bufT + (long long)n * GCN_DH;         // n*128
  float* bufC = bufB + (long long)n * GCN_DH;         // n*128

  const int TB = 256;
  const int nb_nodes  = (n + TB - 1) / TB;
  const int nb_edges  = (E + TB - 1) / TB;
  const int nb_gemm   = (n + 127) / 128;
  const int nb_v128   = (int)(((long long)n * (GCN_DH / 4) + TB - 1) / TB);
  const int nb_v32    = (int)(((long long)n * (GCN_DOUT / 4) + TB - 1) / TB);
  const int nb_edge32 = (int)(((long long)E * 32 + TB - 1) / TB);  // wave per edge
  const int nb_lsm    = (n + 7) / 8;
  const int nb_stats  = 512;
  const long long nvec128 = (long long)n * 32;  // float4 count of [n,128]

  // --- degree normalization ---
  deg_init<<<nb_nodes, TB, 0, stream>>>(dinv, n);
  deg_accum<<<nb_edges, TB, 0, stream>>>(ei, dinv, E);
  deg_to_dinv<<<nb_nodes, TB, 0, stream>>>(dinv, n);

  // --- layer 0: GCNConv(x, W0, b0) -> BN -> ReLU ---
  gcn_gemm_wmma<GCN_DH><<<nb_gemm, 256, 0, stream>>>(x, W0, bufT, n);
  init_out4<GCN_DH><<<nb_v128, TB, 0, stream>>>(bufT, b0, dinv, bufB, n);
  edge_scatter128<<<nb_edge32, TB, 0, stream>>>(bufT, ei, dinv, bufB, E);
  stats_zero<<<1, 256, 0, stream>>>(mean, 256);  // mean & var are adjacent
  bn_partial128<<<nb_stats, 256, 0, stream>>>(bufB, mean, var, nvec128);
  bn_finalize<<<1, 128, 0, stream>>>(mean, var, n, GCN_DH);
  bn_relu4<GCN_DH><<<nb_v128, TB, 0, stream>>>(bufB, mean, var, g0, be0, bufC, n);

  // --- layer 1: GCNConv(h, W1, b1) -> BN -> ReLU ---
  gcn_gemm_wmma<GCN_DH><<<nb_gemm, 256, 0, stream>>>(bufC, W1, bufT, n);
  init_out4<GCN_DH><<<nb_v128, TB, 0, stream>>>(bufT, b1, dinv, bufB, n);
  edge_scatter128<<<nb_edge32, TB, 0, stream>>>(bufT, ei, dinv, bufB, E);
  stats_zero<<<1, 256, 0, stream>>>(mean, 256);
  bn_partial128<<<nb_stats, 256, 0, stream>>>(bufB, mean, var, nvec128);
  bn_finalize<<<1, 128, 0, stream>>>(mean, var, n, GCN_DH);
  bn_relu4<GCN_DH><<<nb_v128, TB, 0, stream>>>(bufB, mean, var, g1, be1, bufC, n);

  // --- layer 2: GCNConv(h, W2, b2) -> log_softmax ---
  gcn_gemm_wmma<GCN_DOUT><<<nb_gemm, 256, 0, stream>>>(bufC, W2, bufT, n);
  init_out4<GCN_DOUT><<<nb_v32, TB, 0, stream>>>(bufT, b2, dinv, bufB, n);
  edge_scatter32<<<nb_edge32, TB, 0, stream>>>(bufT, ei, dinv, bufB, E);
  log_softmax32<<<nb_lsm, 256, 0, stream>>>(bufB, out, n);
}
